// KmeansQuantizer_56513179681190
// MI455X (gfx1250) — compile-verified, matrix operand order checked by reading, not run on hardware
//
#include <hip/hip_runtime.h>
#include <hip/hip_bf16.h>

// Problem constants (from reference)
#define N_TOK   32768           // B*L
#define DIM     768             // D
#define KCODES  4096            // K
#define BETA    0.25f

// Output layout (flat, float32): xq | idx | loss
#define XQ_OFF   0
#define IDX_OFF  ((long)N_TOK * DIM)          // 25165824
#define LOSS_OFF (IDX_OFF + N_TOK)            // 25198592

// Tiling
#define ROWS_WG  128            // rows of x per workgroup (8 waves x 16)
#define BLOCK    256
#define COLT     16             // codes per column tile
#define NTILES   (KCODES / COLT)
#define LDSTR    (DIM + 8)      // padded LDS row stride (bf16 elems) -> conflict-free b128
#define CHUNKS_PER_THREAD 6     // 16 codes * 1536B / 16B / 256 threads

// Workspace layout: [0..63] loss accumulator (+pad) | bf16 codebook | f32 norms
#define WS_CB_OFF   64
#define WS_CN_OFF   (WS_CB_OFF + (size_t)KCODES * DIM * 2)
#define WS_NEED     (WS_CN_OFF + (size_t)KCODES * 4)

#if defined(__has_builtin)
#  if __has_builtin(__builtin_amdgcn_global_load_async_to_lds_b128) && \
      __has_builtin(__builtin_amdgcn_s_wait_asynccnt)
#    define USE_ASYNC 1
#  endif
#endif
#ifndef USE_ASYNC
#  define USE_ASYNC 0
#endif

typedef __attribute__((ext_vector_type(16))) __bf16   v16bf;
typedef __attribute__((ext_vector_type(8)))  float    v8f;
typedef __attribute__((ext_vector_type(4)))  float    vf4;
typedef __attribute__((ext_vector_type(2)))  unsigned vu2;
typedef __attribute__((ext_vector_type(4)))  int      vi4;

#if USE_ASYNC
typedef __attribute__((address_space(1))) void  as1_void;
typedef __attribute__((address_space(3))) void  as3_void;
typedef __attribute__((address_space(1))) vi4   as1_vi4;
typedef __attribute__((address_space(3))) vi4   as3_vi4;
#endif

union Frag {
    v16bf v;
    vf4   f[2];
};

__device__ __forceinline__ unsigned pack_bf16x2(float a, float b) {
    __bf16 ba = (__bf16)a;
    __bf16 bb = (__bf16)b;
    unsigned short ua = __builtin_bit_cast(unsigned short, ba);
    unsigned short ub = __builtin_bit_cast(unsigned short, bb);
    return (unsigned)ua | ((unsigned)ub << 16);
}

__global__ void vq_zero_ws(float* ws) {
    ws[0] = 0.0f;
}

// ------------------------- prep: codebook f32 -> bf16 -------------------------
__global__ void vq_cb_convert(const float* __restrict__ cb, unsigned* __restrict__ cbh2) {
    long i = (long)blockIdx.x * BLOCK + threadIdx.x;   // one per 2 elements
    const long n2 = (long)KCODES * DIM / 2;
    if (i < n2) {
        cbh2[i] = pack_bf16x2(cb[2 * i], cb[2 * i + 1]);
    }
}

// ------------------------- prep: per-code squared norms -----------------------
__global__ void vq_cb_norms(const float* __restrict__ cb, float* __restrict__ cn) {
    const int k   = blockIdx.x;
    const int tid = threadIdx.x;
    const float* row = cb + (long)k * DIM;
    float ss = 0.0f;
    for (int d = tid; d < DIM; d += BLOCK) {
        float v = row[d];
        ss += v * v;
    }
    ss += __shfl_xor(ss, 16, 32);
    ss += __shfl_xor(ss, 8, 32);
    ss += __shfl_xor(ss, 4, 32);
    ss += __shfl_xor(ss, 2, 32);
    ss += __shfl_xor(ss, 1, 32);
    __shared__ float red[BLOCK / 32];
    if ((tid & 31) == 0) red[tid >> 5] = ss;
    __syncthreads();
    if (tid == 0) {
        float s = 0.0f;
#pragma unroll
        for (int w = 0; w < BLOCK / 32; ++w) s += red[w];
        cn[k] = s;
    }
}

// Issue the copy of one 16-code bf16 tile into an LDS buffer.
// 16 codes * 1536 B = 1536 x 16B chunks; 6 per thread. Row-padded LDS dst.
__device__ __forceinline__ void stage_tile(const __bf16* __restrict__ cbh,
                                           __bf16* __restrict__ dst,
                                           int code0, int tid) {
#pragma unroll
    for (int q = 0; q < CHUNKS_PER_THREAD; ++q) {
        int ch = tid + q * BLOCK;            // 0..1535
        int c  = ch / 96;                    // 96 x 16B chunks per 768-elem row
        int wo = (ch - c * 96) * 16;         // byte offset within row
        const char* g = (const char*)(cbh + (size_t)(code0 + c) * DIM) + wo;
        char*       l = (char*)dst + (size_t)c * LDSTR * 2 + wo;
#if USE_ASYNC
        as1_void* gv = (as1_void*)g;
        as3_void* lv = (as3_void*)l;
        __builtin_amdgcn_global_load_async_to_lds_b128(
            (as1_vi4*)gv, (as3_vi4*)lv, 0, 0);
#else
        *(vf4*)(void*)l = *(const vf4*)(const void*)g;   // 16B sync copy fallback
#endif
    }
}

// ---------------- main: bf16 WMMA distance GEMM + argmax (async path) ---------
__global__ void vq_argmax_async_kernel(const float* __restrict__ x,
                                       const __bf16* __restrict__ cbh,
                                       const float* __restrict__ cn_g,
                                       float* __restrict__ out) {
    extern __shared__ char smem[];
    __bf16* lds_x  = (__bf16*)smem;                        // ROWS_WG x LDSTR
    __bf16* lds_b0 = lds_x + (size_t)ROWS_WG * LDSTR;      // COLT x LDSTR
    __bf16* lds_b1 = lds_b0 + (size_t)COLT * LDSTR;        // COLT x LDSTR
    float*  lds_cn = (float*)(lds_b1 + (size_t)COLT * LDSTR);  // KCODES floats

    const int tid  = threadIdx.x;
    const int wave = tid >> 5;
    const int lane = tid & 31;
    const int l15  = lane & 15;
    const int kh   = lane >> 4;
    const long row0 = (long)blockIdx.x * ROWS_WG;

    // Stage x tile: f32 -> bf16 into LDS
    {
        const vf4* x4 = (const vf4*)(x + row0 * DIM);
        for (int i = tid; i < ROWS_WG * (DIM / 4); i += BLOCK) {
            int r  = i / (DIM / 4);
            int c4 = i - r * (DIM / 4);
            vf4 v = x4[i];
            vu2 p;
            p.x = pack_bf16x2(v.x, v.y);
            p.y = pack_bf16x2(v.z, v.w);
            *(vu2*)((void*)(lds_x + (size_t)r * LDSTR + c4 * 4)) = p;
        }
    }
    // Stage all code norms once (16KB)
    for (int i = tid; i < KCODES / 4; i += BLOCK) {
        ((vf4*)lds_cn)[i] = ((const vf4*)cn_g)[i];
    }

    // Prefetch tile 0
    stage_tile(cbh, lds_b0, 0, tid);
    __syncthreads();

    float best[8];
    int   bidx[8];
#pragma unroll
    for (int v = 0; v < 8; ++v) { best[v] = -INFINITY; bidx[v] = 0; }

    for (int t = 0; t < NTILES; ++t) {
        __bf16* cur = (t & 1) ? lds_b1 : lds_b0;
        __bf16* nxt = (t & 1) ? lds_b0 : lds_b1;

        if (t + 1 < NTILES) {
            stage_tile(cbh, nxt, (t + 1) * COLT, tid);   // prefetch next tile
#if USE_ASYNC
            __builtin_amdgcn_s_wait_asynccnt(CHUNKS_PER_THREAD);  // tile t landed
#endif
        } else {
#if USE_ASYNC
            __builtin_amdgcn_s_wait_asynccnt(0);
#endif
        }
        __syncthreads();

        // 16x16 score tile: 24 x wmma_f32_16x16x32_bf16 over D=768
        v8f acc = {0.f, 0.f, 0.f, 0.f, 0.f, 0.f, 0.f, 0.f};
        const __bf16* arow = lds_x + (size_t)(wave * 16 + l15) * LDSTR + kh * 8;
        const __bf16* brow = cur + (size_t)l15 * LDSTR + kh * 8;
#pragma unroll
        for (int k0 = 0; k0 < DIM; k0 += 32) {
            Frag a, b;
            a.f[0] = *(const vf4*)(arow + k0);
            a.f[1] = *(const vf4*)(arow + k0 + 16);
            b.f[0] = *(const vf4*)(brow + k0);
            b.f[1] = *(const vf4*)(brow + k0 + 16);
            acc = __builtin_amdgcn_wmma_f32_16x16x32_bf16(
                false, a.v, false, b.v, (short)0, acc, false, false);
        }

        const float cn   = lds_cn[t * COLT + l15];
        const int   nidx = t * COLT + l15;
#pragma unroll
        for (int v = 0; v < 8; ++v) {
            float s = 2.0f * acc[v] - cn;   // -|x|^2 omitted: constant per row
            if (s > best[v]) { best[v] = s; bidx[v] = nidx; }
        }
        __syncthreads();   // all waves done reading `cur` before it is refilled
    }

    // Cross-lane argmax within each 16-lane half (rows v + 8*kh)
#pragma unroll
    for (int v = 0; v < 8; ++v) {
        float bv = best[v];
        int   bi = bidx[v];
#pragma unroll
        for (int m = 1; m < 16; m <<= 1) {
            float ov = __shfl_xor(bv, m, 32);
            int   oi = __shfl_xor(bi, m, 32);
            if (ov > bv || (ov == bv && oi < bi)) { bv = ov; bi = oi; }
        }
        best[v] = bv;
        bidx[v] = bi;
    }
    if (l15 == 0) {
        float* idx_out = out + IDX_OFF + row0 + wave * 16 + kh * 8;
#pragma unroll
        for (int v = 0; v < 8; ++v) idx_out[v] = (float)bidx[v];
    }
}

// ---------------- fallback main kernel (no workspace codebook) ----------------
__global__ void vq_argmax_kernel(const float* __restrict__ x,
                                 const float* __restrict__ cb,
                                 float* __restrict__ out) {
    extern __shared__ char smem[];
    __bf16* lds_x  = (__bf16*)smem;
    __bf16* lds_b  = (__bf16*)(smem + (size_t)ROWS_WG * LDSTR * 2);
    float*  lds_cn = (float*)(smem + (size_t)(ROWS_WG + COLT) * LDSTR * 2);

    const int tid  = threadIdx.x;
    const int wave = tid >> 5;
    const int lane = tid & 31;
    const int l15  = lane & 15;
    const int kh   = lane >> 4;
    const long row0 = (long)blockIdx.x * ROWS_WG;

    {
        const vf4* x4 = (const vf4*)(x + row0 * DIM);
        for (int i = tid; i < ROWS_WG * (DIM / 4); i += BLOCK) {
            int r  = i / (DIM / 4);
            int c4 = i - r * (DIM / 4);
            vf4 v = x4[i];
            vu2 p;
            p.x = pack_bf16x2(v.x, v.y);
            p.y = pack_bf16x2(v.z, v.w);
            *(vu2*)((void*)(lds_x + (size_t)r * LDSTR + c4 * 4)) = p;
        }
    }
    __syncthreads();

    float best[8];
    int   bidx[8];
#pragma unroll
    for (int v = 0; v < 8; ++v) { best[v] = -INFINITY; bidx[v] = 0; }

    for (int n0 = 0; n0 < KCODES; n0 += COLT) {
        {
            const int c = tid >> 4;
            const int j = tid & 15;
            const vf4* crow = (const vf4*)(cb + (long)(n0 + c) * DIM) + j * 12;
            __bf16* dst = lds_b + (size_t)c * LDSTR + j * 48;
            float ss = 0.0f;
#pragma unroll
            for (int q = 0; q < 12; ++q) {
                vf4 v = crow[q];
                ss += v.x * v.x + v.y * v.y + v.z * v.z + v.w * v.w;
                vu2 p;
                p.x = pack_bf16x2(v.x, v.y);
                p.y = pack_bf16x2(v.z, v.w);
                *(vu2*)((void*)(dst + q * 4)) = p;
            }
            ss += __shfl_xor(ss, 8, 32);
            ss += __shfl_xor(ss, 4, 32);
            ss += __shfl_xor(ss, 2, 32);
            ss += __shfl_xor(ss, 1, 32);
            if (j == 0) lds_cn[c] = ss;
        }
        __syncthreads();

        v8f acc = {0.f, 0.f, 0.f, 0.f, 0.f, 0.f, 0.f, 0.f};
        const __bf16* arow = lds_x + (size_t)(wave * 16 + l15) * LDSTR + kh * 8;
        const __bf16* brow = lds_b + (size_t)l15 * LDSTR + kh * 8;
#pragma unroll
        for (int k0 = 0; k0 < DIM; k0 += 32) {
            Frag a, b;
            a.f[0] = *(const vf4*)(arow + k0);
            a.f[1] = *(const vf4*)(arow + k0 + 16);
            b.f[0] = *(const vf4*)(brow + k0);
            b.f[1] = *(const vf4*)(brow + k0 + 16);
            acc = __builtin_amdgcn_wmma_f32_16x16x32_bf16(
                false, a.v, false, b.v, (short)0, acc, false, false);
        }

        const float cn   = lds_cn[l15];
        const int   nidx = n0 + l15;
#pragma unroll
        for (int v = 0; v < 8; ++v) {
            float s = 2.0f * acc[v] - cn;
            if (s > best[v]) { best[v] = s; bidx[v] = nidx; }
        }
        __syncthreads();
    }

#pragma unroll
    for (int v = 0; v < 8; ++v) {
        float bv = best[v];
        int   bi = bidx[v];
#pragma unroll
        for (int m = 1; m < 16; m <<= 1) {
            float ov = __shfl_xor(bv, m, 32);
            int   oi = __shfl_xor(bi, m, 32);
            if (ov > bv || (ov == bv && oi < bi)) { bv = ov; bi = oi; }
        }
        best[v] = bv;
        bidx[v] = bi;
    }
    if (l15 == 0) {
        float* idx_out = out + IDX_OFF + row0 + wave * 16 + kh * 8;
#pragma unroll
        for (int v = 0; v < 8; ++v) idx_out[v] = (float)bidx[v];
    }
}

// Gather xq = codebook[idx] and accumulate sum((q - x)^2) into ws[0].
__global__ void vq_gather_loss_kernel(const float* __restrict__ x,
                                      const float* __restrict__ cb,
                                      float* __restrict__ out,
                                      float* __restrict__ loss_acc) {
    const int row = blockIdx.x;
    const int tid = threadIdx.x;
    const int idx = (int)(out[IDX_OFF + row] + 0.5f);

    const float* q  = cb + (long)idx * DIM;
    const float* xr = x + (long)row * DIM;
    float*       xo = out + (long)row * DIM;

    float ss = 0.0f;
    for (int d = tid; d < DIM; d += BLOCK) {
        float qv = q[d];
        float xv = xr[d];
        xo[d] = qv;                       // xq == q numerically (STE)
        float df = qv - xv;
        ss += df * df;
    }
    ss += __shfl_xor(ss, 16, 32);
    ss += __shfl_xor(ss, 8, 32);
    ss += __shfl_xor(ss, 4, 32);
    ss += __shfl_xor(ss, 2, 32);
    ss += __shfl_xor(ss, 1, 32);

    __shared__ float red[BLOCK / 32];
    if ((tid & 31) == 0) red[tid >> 5] = ss;
    __syncthreads();
    if (tid == 0) {
        float s = 0.0f;
#pragma unroll
        for (int w = 0; w < BLOCK / 32; ++w) s += red[w];
        atomicAdd(loss_acc, s);
    }
}

__global__ void vq_loss_finalize(const float* __restrict__ loss_acc,
                                 float* __restrict__ out) {
    out[LOSS_OFF] = (1.0f + BETA) * loss_acc[0] / (float)((long)N_TOK * DIM);
}

extern "C" void kernel_launch(void* const* d_in, const int* in_sizes, int n_in,
                              void* d_out, int out_size, void* d_ws, size_t ws_size,
                              hipStream_t stream) {
    const float* x  = (const float*)d_in[0];   // [B,L,D] f32
    const float* cb = (const float*)d_in[1];   // [K,D]   f32
    float*   out  = (float*)d_out;
    float*   lacc = (float*)d_ws;
    __bf16*  cbh  = (__bf16*)((char*)d_ws + WS_CB_OFF);
    float*   cng  = (float*)((char*)d_ws + WS_CN_OFF);

    vq_zero_ws<<<1, 1, 0, stream>>>(lacc);

    if (ws_size >= WS_NEED) {
        vq_cb_convert<<<(int)(((long)KCODES * DIM / 2 + BLOCK - 1) / BLOCK), BLOCK, 0, stream>>>(
            cb, (unsigned*)cbh);
        vq_cb_norms<<<KCODES, BLOCK, 0, stream>>>(cb, cng);

        const size_t lds_a = (size_t)(ROWS_WG + 2 * COLT) * LDSTR * 2 + (size_t)KCODES * 4;
        (void)hipFuncSetAttribute((const void*)vq_argmax_async_kernel,
                                  hipFuncAttributeMaxDynamicSharedMemorySize, (int)lds_a);
        vq_argmax_async_kernel<<<N_TOK / ROWS_WG, BLOCK, lds_a, stream>>>(x, cbh, cng, out);
    } else {
        const size_t lds_b = (size_t)(ROWS_WG + COLT) * LDSTR * 2 + COLT * sizeof(float);
        (void)hipFuncSetAttribute((const void*)vq_argmax_kernel,
                                  hipFuncAttributeMaxDynamicSharedMemorySize, (int)lds_b);
        vq_argmax_kernel<<<N_TOK / ROWS_WG, BLOCK, lds_b, stream>>>(x, cb, out);
    }

    vq_gather_loss_kernel<<<N_TOK, BLOCK, 0, stream>>>(x, cb, out, lacc);
    vq_loss_finalize<<<1, 1, 0, stream>>>(lacc, out);
}
